// CVX_Reasoning_Engine_47029891891832
// MI455X (gfx1250) — matrix operand order repaired and checked by reading, not compiled, not run on hardware
//
#include <hip/hip_runtime.h>
#include <hip/hip_bf16.h>

// ---------------------------------------------------------------------------
// CDNA5 (gfx1250) fused MLP + QP solver.
// MLP layers run as bf16 WMMA GEMMs (v_wmma_f32_16x16x32_bf16, f32 accum),
// with block-level LDS staging of the weight tiles (double-buffered).
// K/N are template constants -> fully unrolled, branch-free inner pipeline.
// ---------------------------------------------------------------------------

typedef __bf16 bf16_t;
typedef __attribute__((ext_vector_type(16))) __bf16 v16bf;
typedef __attribute__((ext_vector_type(8)))  __bf16 v8bf;
typedef __attribute__((ext_vector_type(8)))  float  v8f;

#define BS    16384
#define ZDIM  512
#define NOBJ  64

#define LDS_STRIDE 40   // padded row stride (elements) for the 64x32 B tile

// ---------------------------------------------------------------------------
// Prep kernels
// ---------------------------------------------------------------------------

__global__ void cast_to_bf16(const float* __restrict__ x, bf16_t* __restrict__ y, int n) {
    int i = blockIdx.x * blockDim.x + threadIdx.x;
    if (i < n) y[i] = (bf16_t)x[i];
}

// WT[n*K + k] = bf16( W[k*N + n] )   (W row stride = N; only first K rows used)
__global__ void transpose_to_bf16(const float* __restrict__ W, bf16_t* __restrict__ WT,
                                  int K, int N) {
    int idx = blockIdx.x * blockDim.x + threadIdx.x;
    if (idx >= K * N) return;
    int n = idx / K;
    int k = idx - n * K;
    WT[idx] = (bf16_t)W[(size_t)k * N + n];
}

// Fold constant bounds rows of W1 into layer-1 bias: c1p[n] = c1[n] + sum_j b[j]*W1[512+j][n]
__global__ void fold_bias1(const float* __restrict__ W1, const float* __restrict__ c1,
                           const float* __restrict__ bounds, float* __restrict__ c1p, int N) {
    int n = blockIdx.x * blockDim.x + threadIdx.x;
    if (n >= N) return;
    float v = c1[n];
#pragma unroll
    for (int j = 0; j < 4; ++j)
        v += bounds[j] * W1[(size_t)(ZDIM + j) * N + n];
    c1p[n] = v;
}

// ---------------------------------------------------------------------------
// WMMA GEMM:  C = act( A(MxK,bf16,row) @ WT^T + bias ),  WT is N x K bf16 row-major.
// Block = 256 threads = 8 waves. Block tile: 256 rows x 64 cols.
// Each wave: 32 rows x 64 cols = 8 accumulators; B tile staged in LDS
// (double-buffered, 64 cols x 32 k, cooperative load: 1 b128 per thread).
// ---------------------------------------------------------------------------
template <int K, int N, bool LRELU, typename OutT>
__global__ __launch_bounds__(256)
void mlp_gemm_wmma(const bf16_t* __restrict__ A,
                   const bf16_t* __restrict__ WT,
                   const float*  __restrict__ bias,
                   OutT*         __restrict__ C) {
    __shared__ bf16_t Bs[2][64 * LDS_STRIDE];

    const int tid  = threadIdx.x;
    const int lane = tid & 31;
    const int wave = tid >> 5;
    const int half = lane >> 4;    // 0: lanes 0-15, 1: lanes 16-31
    const int l16  = lane & 15;
    const int row0 = blockIdx.x * 256 + wave * 32;
    const int col0 = blockIdx.y * 64;

    // Cooperative B staging: thread t covers LDS row n = t>>2, 8 bf16 at (t&3)*8.
    const int sn = tid >> 2;
    const int sk = (tid & 3) * 8;
    const bf16_t* Wst = WT + (size_t)(col0 + sn) * K + sk;

    v8f acc[8] = {};

    const bf16_t* Arow0 = A + (size_t)(row0 + l16) * K + 16 * half;
    const bf16_t* Arow1 = Arow0 + (size_t)16 * K;

    // Prologue: stage k0 = 0.
    *(v8bf*)(&Bs[0][sn * LDS_STRIDE + sk]) = *(const v8bf*)(Wst);
    __syncthreads();

#pragma unroll
    for (int k0 = 0; k0 < K; k0 += 32) {
        const int buf = (k0 >> 5) & 1;
        // Prefetch next B k-slice into the other buffer (overlaps compute).
        if (k0 + 32 < K)
            *(v8bf*)(&Bs[buf ^ 1][sn * LDS_STRIDE + sk]) = *(const v8bf*)(Wst + k0 + 32);

        // A fragments (16x32 bf16): lane<16 holds K {k0..k0+15},
        // lane>=16 holds K {k0+16..k0+31}.
        v8bf a0lo = *(const v8bf*)(Arow0 + k0);
        v8bf a0hi = *(const v8bf*)(Arow0 + k0 + 8);
        v8bf a1lo = *(const v8bf*)(Arow1 + k0);
        v8bf a1hi = *(const v8bf*)(Arow1 + k0 + 8);
        v16bf a0, a1;
#pragma unroll
        for (int i = 0; i < 8; ++i) {
            a0[i] = a0lo[i]; a0[8 + i] = a0hi[i];
            a1[i] = a1lo[i]; a1[8 + i] = a1hi[i];
        }

#pragma unroll
        for (int nt = 0; nt < 4; ++nt) {
            // B fragment: lane holds one output column, 16 contiguous K per half-wave.
            v16bf b = *(const v16bf*)(&Bs[buf][(nt * 16 + l16) * LDS_STRIDE + 16 * half]);
            acc[nt]     = __builtin_amdgcn_wmma_f32_16x16x32_bf16(
                false, a0, false, b, (short)0, acc[nt],     false, false);
            acc[4 + nt] = __builtin_amdgcn_wmma_f32_16x16x32_bf16(
                false, a1, false, b, (short)0, acc[4 + nt], false, false);
        }
        __syncthreads();
    }

    // Epilogue: C/D layout = lane<16 -> M=j, lane>=16 -> M=8+j; N = col0+nt*16+l16.
#pragma unroll
    for (int rt = 0; rt < 2; ++rt) {
#pragma unroll
        for (int nt = 0; nt < 4; ++nt) {
            const int col = col0 + nt * 16 + l16;
            const float bv = bias[col];
#pragma unroll
            for (int j = 0; j < 8; ++j) {
                const int row = row0 + rt * 16 + half * 8 + j;
                float v = acc[rt * 4 + nt][j] + bv;
                if (LRELU) v = (v > 0.0f) ? v : 0.2f * v;
                C[(size_t)row * N + col] = (OutT)v;
            }
        }
    }
}

// ---------------------------------------------------------------------------
// Exact KKT solve per object/axis (matches _solve_axis in the reference)
// ---------------------------------------------------------------------------
__device__ __forceinline__ void solve_axis(float pp, float pg, float lo, float hi,
                                           float& x, float& w) {
    float x0 = fmaxf(pp, lo);
    float g0 = fmaxf(pg, 1.0f);
    float t  = 0.5f * (hi - pp - pg);
    float xl = fminf(fmaxf(pp + t, lo), hi - 1.0f);
    bool over = (x0 + g0) > hi;
    x = over ? xl : x0;
    w = over ? hi : (x0 + g0);
}

__global__ void qp_kernel(const float* __restrict__ P, const float* __restrict__ bounds,
                          float* __restrict__ out, int total) {
    int i = blockIdx.x * blockDim.x + threadIdx.x;
    if (i >= total) return;
    float4 p = *(const float4*)(P + 4 * (size_t)i);  // [px, py, pw, ph]
    float b0 = bounds[0], b1 = bounds[1], b2 = bounds[2], b3 = bounds[3];
    float xs, ws, ys, hs;
    solve_axis(p.x, p.z, b0, b2, xs, ws);
    solve_axis(p.y, p.w, b1, b3, ys, hs);
    float4 r = make_float4(xs, ys, ws, hs);
    *(float4*)(out + 4 * (size_t)i) = r;
}

// ---------------------------------------------------------------------------
// Launcher
// ---------------------------------------------------------------------------
extern "C" void kernel_launch(void* const* d_in, const int* in_sizes, int n_in,
                              void* d_out, int out_size, void* d_ws, size_t ws_size,
                              hipStream_t stream) {
    // setup_inputs order: z, bounds, W1, c1, W2, c2, W3, c3, W4, c4, W5, c5
    const float* z      = (const float*)d_in[0];
    const float* bounds = (const float*)d_in[1];
    const float* W1 = (const float*)d_in[2];  const float* c1 = (const float*)d_in[3];
    const float* W2 = (const float*)d_in[4];  const float* c2 = (const float*)d_in[5];
    const float* W3 = (const float*)d_in[6];  const float* c3 = (const float*)d_in[7];
    const float* W4 = (const float*)d_in[8];  const float* c4 = (const float*)d_in[9];
    const float* W5 = (const float*)d_in[10]; const float* c5 = (const float*)d_in[11];
    float* out = (float*)d_out;

    // Workspace carve-up
    char* ws = (char*)d_ws;
    bf16_t* Xb  = (bf16_t*)ws; ws += (size_t)BS * 512 * sizeof(bf16_t);
    bf16_t* H1  = (bf16_t*)ws; ws += (size_t)BS * 512 * sizeof(bf16_t);
    bf16_t* H2  = (bf16_t*)ws; ws += (size_t)BS * 256 * sizeof(bf16_t);
    bf16_t* H3  = (bf16_t*)ws; ws += (size_t)BS * 128 * sizeof(bf16_t);
    bf16_t* H4  = (bf16_t*)ws; ws += (size_t)BS * 64  * sizeof(bf16_t);
    float*  P   = (float*)ws;  ws += (size_t)BS * 256 * sizeof(float);
    bf16_t* WT1 = (bf16_t*)ws; ws += (size_t)512 * 512 * sizeof(bf16_t);
    bf16_t* WT2 = (bf16_t*)ws; ws += (size_t)256 * 512 * sizeof(bf16_t);
    bf16_t* WT3 = (bf16_t*)ws; ws += (size_t)128 * 256 * sizeof(bf16_t);
    bf16_t* WT4 = (bf16_t*)ws; ws += (size_t)64  * 128 * sizeof(bf16_t);
    bf16_t* WT5 = (bf16_t*)ws; ws += (size_t)256 * 64  * sizeof(bf16_t);
    float*  c1p = (float*)ws;  ws += (size_t)512 * sizeof(float);

    const int T = 256;
    // --- prep: cast activations, transpose+cast weights, fold bounds into bias1
    {
        int n = BS * 512;
        cast_to_bf16<<<(n + T - 1) / T, T, 0, stream>>>(z, Xb, n);
        int e1 = 512 * 512; transpose_to_bf16<<<(e1 + T - 1) / T, T, 0, stream>>>(W1, WT1, 512, 512);
        int e2 = 512 * 256; transpose_to_bf16<<<(e2 + T - 1) / T, T, 0, stream>>>(W2, WT2, 512, 256);
        int e3 = 256 * 128; transpose_to_bf16<<<(e3 + T - 1) / T, T, 0, stream>>>(W3, WT3, 256, 128);
        int e4 = 128 * 64;  transpose_to_bf16<<<(e4 + T - 1) / T, T, 0, stream>>>(W4, WT4, 128, 64);
        int e5 = 64 * 256;  transpose_to_bf16<<<(e5 + T - 1) / T, T, 0, stream>>>(W5, WT5, 64, 256);
        fold_bias1<<<(512 + T - 1) / T, T, 0, stream>>>(W1, c1, bounds, c1p, 512);
    }

    // --- MLP: 5 WMMA GEMMs (grid.x = M/256 row-blocks, grid.y = N/64 col-blocks)
    dim3 blk(256);
    mlp_gemm_wmma<512, 512, true,  bf16_t><<<dim3(BS / 256, 512 / 64), blk, 0, stream>>>(Xb, WT1, c1p, H1);
    mlp_gemm_wmma<512, 256, true,  bf16_t><<<dim3(BS / 256, 256 / 64), blk, 0, stream>>>(H1, WT2, c2,  H2);
    mlp_gemm_wmma<256, 128, true,  bf16_t><<<dim3(BS / 256, 128 / 64), blk, 0, stream>>>(H2, WT3, c3,  H3);
    mlp_gemm_wmma<128, 64,  true,  bf16_t><<<dim3(BS / 256, 64  / 64), blk, 0, stream>>>(H3, WT4, c4,  H4);
    mlp_gemm_wmma<64,  256, false, float ><<<dim3(BS / 256, 256 / 64), blk, 0, stream>>>(H4, WT5, c5,  P);

    // --- QP epilogue (elementwise, exact KKT)
    int total = BS * NOBJ;
    qp_kernel<<<(total + T - 1) / T, T, 0, stream>>>(P, bounds, out, total);
}
